// ObliviousDecisionTree_86363202388544
// MI455X (gfx1250) — matrix operand order repaired and checked by reading, not compiled, not run on hardware
//
#include <hip/hip_runtime.h>

typedef _Float16 v16h __attribute__((ext_vector_type(16)));
typedef _Float16 v8h  __attribute__((ext_vector_type(8)));
typedef _Float16 h2   __attribute__((ext_vector_type(2)));
typedef float    v8f  __attribute__((ext_vector_type(8)));

#define DIM    256
#define NTREES 512
#define DEPTH  6
#define UNITS  16
#define NLEAF  64
#define BATCH  1024
#define ND     (NTREES*DEPTH)   // 3072

static __device__ __forceinline__ v16h cat8(v8h lo, v8h hi) {
    return __builtin_shufflevector(lo, hi, 0,1,2,3,4,5,6,7,8,9,10,11,12,13,14,15);
}

// ---------------- K0: zero output (d_out is poisoned; K3 accumulates) -------
__global__ void k_zero(float* __restrict__ out, int n) {
    int t = blockIdx.x * blockDim.x + threadIdx.x;
    if (t < n) out[t] = 0.0f;
}

// ---------------- K0b: f32 -> f16 conversion (x, response) ------------------
__global__ void k_cvt(const float* __restrict__ src, _Float16* __restrict__ dst, int n) {
    int t = blockIdx.x * blockDim.x + threadIdx.x;
    if (t < n) dst[t] = (_Float16)src[t];
}

// ---------------- K1: sparsemax over depth; emit selectors TRANSPOSED f16 ---
// selT[(n*6+d)*256 + i]  -- row-major in K(=i) so K2's B operand is two b128s
__global__ void k_sparsemax(const float* __restrict__ logits,
                            _Float16* __restrict__ selT) {
    int t = blockIdx.x * blockDim.x + threadIdx.x;   // 0 .. 256*512-1
    if (t >= DIM * NTREES) return;
    const int i = t / NTREES, n = t % NTREES;        // logits (dim, trees, depth)
    const float* z = logits + (size_t)t * DEPTH;
    float v[DEPTH], s[DEPTH];
#pragma unroll
    for (int d = 0; d < DEPTH; ++d) { v[d] = z[d]; s[d] = v[d]; }
#pragma unroll
    for (int a = 1; a < DEPTH; ++a)                  // descending sort net
#pragma unroll
        for (int b = a; b > 0; --b) {
            float hi = fmaxf(s[b - 1], s[b]);
            float lo = fminf(s[b - 1], s[b]);
            s[b - 1] = hi; s[b] = lo;
        }
    float cs = 0.0f, csk = 1.0f; int k = 1;
#pragma unroll
    for (int j = 0; j < DEPTH; ++j) {
        cs += s[j];
        if (s[j] * (float)(j + 1) > cs - 1.0f) { k = j + 1; csk = cs; }
    }
    float tau = (csk - 1.0f) / (float)k;
#pragma unroll
    for (int d = 0; d < DEPTH; ++d)
        selT[(size_t)(n * DEPTH + d) * DIM + i] = (_Float16)fmaxf(v[d] - tau, 0.0f);
}

// ---------------- K2: fv = x @ sel (pure-WMMA, LDS-free) + threshold/temp ---
// one wave computes a 16x64 tile (4 N-subtiles share the A operand);
// operands are direct b128 f16 loads in contiguous k order (k = 16*hi + j), same for A and B
__global__ void k_fv_tl(const _Float16* __restrict__ x16,
                        const _Float16* __restrict__ selT,
                        const float* __restrict__ thr,
                        const float* __restrict__ logT,
                        float* __restrict__ tl) {
    const int L  = threadIdx.x;      // 0..31
    const int m  = L & 15;
    const int hi = L >> 4;
    const int m0  = (blockIdx.x / (ND / 64)) * 16;
    const int nd0 = (blockIdx.x % (ND / 64)) * 64;

    v8f acc0 = {}, acc1 = {}, acc2 = {}, acc3 = {};
    const _Float16* arow = x16 + (size_t)(m0 + m) * DIM;
    const _Float16* b0r  = selT + (size_t)(nd0 +  0 + m) * DIM;
    const _Float16* b1r  = selT + (size_t)(nd0 + 16 + m) * DIM;
    const _Float16* b2r  = selT + (size_t)(nd0 + 32 + m) * DIM;
    const _Float16* b3r  = selT + (size_t)(nd0 + 48 + m) * DIM;

    for (int k0 = 0; k0 < DIM; k0 += 32) {
        // A layout: element j holds K = k0 + 16*hi + j  (contiguous 16 halves)
        v16h a = cat8(*(const v8h*)(arow + k0 + 16 * hi),
                      *(const v8h*)(arow + k0 + 16 * hi + 8));
        // B layout: element j holds K = k0 + 16*hi + j  (contiguous 16 halves)
        v16h b0 = cat8(*(const v8h*)(b0r + k0 + 16 * hi),
                       *(const v8h*)(b0r + k0 + 16 * hi + 8));
        v16h b1 = cat8(*(const v8h*)(b1r + k0 + 16 * hi),
                       *(const v8h*)(b1r + k0 + 16 * hi + 8));
        v16h b2 = cat8(*(const v8h*)(b2r + k0 + 16 * hi),
                       *(const v8h*)(b2r + k0 + 16 * hi + 8));
        v16h b3 = cat8(*(const v8h*)(b3r + k0 + 16 * hi),
                       *(const v8h*)(b3r + k0 + 16 * hi + 8));
        acc0 = __builtin_amdgcn_wmma_f32_16x16x32_f16(false, a, false, b0, (short)0, acc0, false, false);
        acc1 = __builtin_amdgcn_wmma_f32_16x16x32_f16(false, a, false, b1, (short)0, acc1, false, false);
        acc2 = __builtin_amdgcn_wmma_f32_16x16x32_f16(false, a, false, b2, (short)0, acc2, false, false);
        acc3 = __builtin_amdgcn_wmma_f32_16x16x32_f16(false, a, false, b3, (short)0, acc3, false, false);
    }
    // fused transform + store: tl = (fv - thr) * exp(-logT)
    v8f accs[4] = {acc0, acc1, acc2, acc3};
#pragma unroll
    for (int s = 0; s < 4; ++s) {
        int nd = nd0 + s * 16 + m;                   // N = lane%16; nd == n*6+d
        float sc = __expf(-logT[nd]);
        float th = thr[nd];
#pragma unroll
        for (int r = 0; r < 8; ++r) {
            int brow = m0 + r + (hi << 3);           // C layout: M = r + 8*hi
            tl[(size_t)brow * ND + nd] = (accs[s][r] - th) * sc;
        }
    }
}

// ---------------- K3: leaf products + response GEMM + mean over trees -------
// block = 1 wave: 16 batch rows x 16 trees; leaf tile staged in LDS as f16
__global__ void k_trees(const float* __restrict__ tl,
                        const _Float16* __restrict__ resp16,
                        float* __restrict__ out) {
    __shared__ _Float16 leafH[16 * NLEAF];           // 16 rows x 64 leaves, f16
    const int L  = threadIdx.x;
    const int m  = L & 15;
    const int hi = L >> 4;
    const int b0 = blockIdx.x * 16;
    const int n0 = blockIdx.y * 16;

    v8f acc0 = {}, acc1 = {};                        // independent K-chunks
    for (int nt = 0; nt < 16; ++nt) {
        const int n = n0 + nt;
        if (nt + 1 < 16)   // global_prefetch_b8 of next tree's response rows
            __builtin_prefetch(resp16 + (size_t)(n + 1) * UNITS * NLEAF + m * NLEAF, 0, 1);

        // sparsemoid gates for this lane's batch row
        const float* trow = tl + (size_t)(b0 + m) * ND + n * DEPTH;
        float g1[DEPTH];
#pragma unroll
        for (int d = 0; d < DEPTH; ++d)
            g1[d] = fminf(fmaxf(0.5f + 0.5f * trow[d], 0.0f), 1.0f);

        // leaf doubling over depths 0..4 -> 32 partial products (registers)
        float v[32];
        {
            float gl, gh;
            gl = g1[0]; gh = 1.0f - gl;
            v[0] = gl; v[1] = gh;
            gl = g1[1]; gh = 1.0f - gl;
            v[2] = v[0] * gh; v[3] = v[1] * gh; v[0] *= gl; v[1] *= gl;
            gl = g1[2]; gh = 1.0f - gl;
#pragma unroll
            for (int c = 0; c < 4; ++c)  { v[4 + c]  = v[c] * gh; v[c] *= gl; }
            gl = g1[3]; gh = 1.0f - gl;
#pragma unroll
            for (int c = 0; c < 8; ++c)  { v[8 + c]  = v[c] * gh; v[c] *= gl; }
            gl = g1[4]; gh = 1.0f - gl;
#pragma unroll
            for (int c = 0; c < 16; ++c) { v[16 + c] = v[c] * gh; v[c] *= gl; }
        }
        // depth-5 factor splits leaves between the half-waves; store f16 pairs
        float f5 = hi ? (1.0f - g1[5]) : g1[5];
        _Float16* lrow = leafH + m * NLEAF + hi * 32;
#pragma unroll
        for (int c = 0; c < 16; ++c) {
            h2 pr; pr[0] = (_Float16)(v[2 * c] * f5);
                   pr[1] = (_Float16)(v[2 * c + 1] * f5);
            *(h2*)(lrow + 2 * c) = pr;
        }
        __syncthreads();

        // A gather: consecutive-K f16 pairs from LDS, contiguous k order (16*hi + j) matching B
        v16h a0, a1;
#pragma unroll
        for (int p = 0; p < 8; ++p) {
            int cofs = (p << 1) + (hi << 4);
            h2 p0 = *(const h2*)(leafH + m * NLEAF + cofs);
            h2 p1 = *(const h2*)(leafH + m * NLEAF + 32 + cofs);
            a0[2 * p] = p0[0]; a0[2 * p + 1] = p0[1];
            a1[2 * p] = p1[0]; a1[2 * p + 1] = p1[1];
        }
        // B: response[n] rows, already f16, contiguous in K -> aligned b128s
        const _Float16* rrow = resp16 + (size_t)n * (UNITS * NLEAF) + m * NLEAF + 16 * hi;
        v16h bb0 = cat8(*(const v8h*)(rrow),      *(const v8h*)(rrow + 8));
        v16h bb1 = cat8(*(const v8h*)(rrow + 32), *(const v8h*)(rrow + 40));

        acc0 = __builtin_amdgcn_wmma_f32_16x16x32_f16(false, a0, false, bb0, (short)0, acc0, false, false);
        acc1 = __builtin_amdgcn_wmma_f32_16x16x32_f16(false, a1, false, bb1, (short)0, acc1, false, false);
        __syncthreads();
    }
    // mean over 512 trees, distributed across gridDim.y tree-groups
    const float scale = 1.0f / (float)NTREES;
#pragma unroll
    for (int r = 0; r < 8; ++r) {
        int brow = b0 + r + (hi << 3);
        atomicAdd(&out[brow * UNITS + m], (acc0[r] + acc1[r]) * scale);
    }
}

// ---------------------------------------------------------------------------
extern "C" void kernel_launch(void* const* d_in, const int* in_sizes, int n_in,
                              void* d_out, int out_size, void* d_ws, size_t ws_size,
                              hipStream_t stream) {
    const float* x    = (const float*)d_in[0];  // (1024, 256)
    const float* fsl  = (const float*)d_in[1];  // (256, 512, 6)
    const float* thr  = (const float*)d_in[2];  // (512, 6)
    const float* logT = (const float*)d_in[3];  // (512, 6)
    const float* resp = (const float*)d_in[4];  // (512, 16, 64)
    float* out = (float*)d_out;                 // (1024, 16)

    // workspace carve-up (all offsets 16B-aligned)
    char* ws = (char*)d_ws;
    _Float16* x16    = (_Float16*)(ws);                               // 512 KB
    _Float16* selT   = (_Float16*)(ws + 524288);                      // 1.5 MB
    _Float16* resp16 = (_Float16*)(ws + 524288 + 1572864);            // 1 MB
    float*    tl     = (float*)   (ws + 524288 + 1572864 + 1048576);  // 12.6 MB

    k_zero<<<(BATCH * UNITS + 255) / 256, 256, 0, stream>>>(out, BATCH * UNITS);
    k_cvt<<<(BATCH * DIM + 255) / 256, 256, 0, stream>>>(x, x16, BATCH * DIM);
    k_cvt<<<(NTREES * UNITS * NLEAF + 255) / 256, 256, 0, stream>>>(
        resp, resp16, NTREES * UNITS * NLEAF);
    k_sparsemax<<<(DIM * NTREES + 255) / 256, 256, 0, stream>>>(fsl, selT);
    k_fv_tl<<<(BATCH / 16) * (ND / 64), 32, 0, stream>>>(x16, selT, thr, logT, tl);
    k_trees<<<dim3(BATCH / 16, NTREES / 16), 32, 0, stream>>>(tl, resp16, out);
}
